// Attention_63101659512856
// MI455X (gfx1250) — compile-verified, matrix operand order checked by reading, not run on hardware
//
#include <hip/hip_runtime.h>

typedef _Float16 half_t;
typedef __attribute__((ext_vector_type(16))) _Float16 v16h;
typedef __attribute__((ext_vector_type(8)))  float    v8f;
typedef unsigned int u32x4 __attribute__((ext_vector_type(4)));
typedef int          i32x4 __attribute__((ext_vector_type(4)));
typedef int          i32x8 __attribute__((ext_vector_type(8)));

#define SEQ 2048
#define DIM 4096
#define NH  32
#define NKV 8
#define HD  128

// ---------------------------------------------------------------------------
// LDS byte offset of a __shared__ object (flat shared addr keeps the LDS
// offset in the low 32 bits).
// ---------------------------------------------------------------------------
__device__ __forceinline__ unsigned lds_off_of(const void* p) {
  return (unsigned)(unsigned long long)p;
}

// ---------------------------------------------------------------------------
// Tensor Data Mover: 2D tile load global->LDS.
// D# packing per CDNA5 ISA (§8.3/§8.4):
//  g0: [1:0]=count=1, [63:32]=lds_addr, [120:64]=global_addr, [127:126]=type=2
//  g1: [17:16]=data_size(2B), [79:48]=tensor_dim0, [111:80]=tensor_dim1,
//      [127:112]=tile_dim0, [143:128]=tile_dim1, [207:160]=tensor_dim0_stride
// ---------------------------------------------------------------------------
__device__ __forceinline__ void tdm_load_2d(unsigned lds_off, const void* gptr,
                                            int rows, int k_elems,
                                            int tile_rows, int tile_cols) {
  unsigned long long ga = (unsigned long long)gptr;
  u32x4 g0;
  g0.x = 1u;                                            // count=1, user mode
  g0.y = lds_off;                                       // lds_addr
  g0.z = (unsigned)ga;                                  // global_addr[31:0]
  g0.w = (unsigned)((ga >> 32) & 0x1FFFFFFu) | (2u << 30);  // addr hi | type=2
  i32x8 g1;
  g1[0] = (1 << 16);                                    // data_size = 2 bytes
  g1[1] = (k_elems & 0xFFFF) << 16;                     // tensor_dim0 lo
  g1[2] = ((k_elems >> 16) & 0xFFFF) | ((rows & 0xFFFF) << 16);
  g1[3] = ((rows >> 16) & 0xFFFF) | (tile_cols << 16);  // dim1 hi | tile_dim0
  g1[4] = tile_rows;                                    // tile_dim1, tile_dim2=0
  g1[5] = k_elems;                                      // dim0_stride[31:0]
  g1[6] = 0;
  g1[7] = 0;
  i32x4 z4 = {0, 0, 0, 0};
#if __clang_major__ >= 23
  i32x8 z8 = {0, 0, 0, 0, 0, 0, 0, 0};
  __builtin_amdgcn_tensor_load_to_lds(g0, g1, z4, z4, z8, 0);
#else
  __builtin_amdgcn_tensor_load_to_lds(g0, g1, z4, z4, 0);
#endif
}

// ---------------------------------------------------------------------------
// BFP block-quantize (group = 64 columns across all rows) -> f16 weights.
// ---------------------------------------------------------------------------
__global__ __launch_bounds__(256)
void bfp_quant_to_f16(const float* __restrict__ W, half_t* __restrict__ Wh,
                      int rows, int cols) {
  const int j = blockIdx.x;
  const int tid = threadIdx.x;
  __shared__ float red[256];
  __shared__ float s_bias, s_gexp;
  const int n = rows * 64;

  float mx = 0.f;
  for (int i = tid; i < n; i += 256) {
    int r = i >> 6, c = i & 63;
    mx = fmaxf(mx, fabsf(W[(size_t)r * cols + j * 64 + c]));
  }
  red[tid] = mx;
  __syncthreads();
  for (int s = 128; s > 0; s >>= 1) {
    if (tid < s) red[tid] = fmaxf(red[tid], red[tid + s]);
    __syncthreads();
  }
  if (tid == 0) {
    float maxv = red[0];
    float bias = 256.0f - log2f(maxv) + log2f(2.0f - exp2f(-7.0f)) - 1.0f;
    float max_exp = 255.0f - bias;
    float max_val = exp2f(max_exp) * (2.0f - exp2f(-9.0f));
    float gexp = floorf(log2f(fminf(maxv, max_val) + 1e-40f));
    if (gexp + bias < 1.0f) gexp = -bias;
    s_bias = bias;
    s_gexp = gexp;
  }
  __syncthreads();
  const float bias = s_bias;
  const float gexp = s_gexp;
  const float max_val = exp2f(255.0f - bias) * (2.0f - exp2f(-9.0f));
  const float g_scale = exp2f(gexp);

  for (int i = tid; i < n; i += 256) {
    int r = i >> 6, c = i & 63;
    size_t idx = (size_t)r * cols + j * 64 + c;
    float w = W[idx];
    float sign = (w > 0.f) ? 1.f : ((w < 0.f) ? -1.f : 0.f);
    float absx = fminf(fabsf(w), max_val);
    float expo = floorf(log2f(absx + 1e-40f));
    float mant = absx * exp2f(-expo);
    int sm = (int)(mant * 128.0f);
    int shift = (int)(gexp - expo);
    shift = min(max(shift, 0), 31);
    int shifted = sm >> shift;
    int safe = max(shifted, 1);
    int bitlen = (int)floorf(log2f((float)safe)) + 1;
    int mask = (bitlen > 8) ? (255 << (bitlen - 8)) : 255;
    float trunc = (float)(shifted & mask) * (1.0f / 128.0f);
    Wh[idx] = (half_t)(sign * trunc * g_scale);
  }
}

// ---------------------------------------------------------------------------
__global__ __launch_bounds__(256)
void f32_to_f16(const float* __restrict__ in, half_t* __restrict__ out, int n) {
  int i = (blockIdx.x * 256 + threadIdx.x) * 4;
  if (i >= n) return;
  float4 v = *(const float4*)(in + i);
  union { half_t h[4]; unsigned long long u; } pk;
  pk.h[0] = (half_t)v.x; pk.h[1] = (half_t)v.y;
  pk.h[2] = (half_t)v.z; pk.h[3] = (half_t)v.w;
  *(unsigned long long*)(out + i) = pk.u;
}

// ---------------------------------------------------------------------------
__global__ __launch_bounds__(256)
void rope_kernel(half_t* __restrict__ t, const float* __restrict__ cb,
                 const float* __restrict__ sb, int heads) {
  int i = blockIdx.x * 256 + threadIdx.x;
  int total = SEQ * heads * (HD / 2);
  if (i >= total) return;
  int p = i % (HD / 2);
  int h = (i / (HD / 2)) % heads;
  int s = i / ((HD / 2) * heads);
  size_t base = ((size_t)s * heads + h) * HD + 2 * p;
  float te = (float)t[base], to = (float)t[base + 1];
  float c = cb[s * (HD / 2) + p], sn = sb[s * (HD / 2) + p];
  t[base]     = (half_t)(te * c - to * sn);
  t[base + 1] = (half_t)(te * sn + to * c);
}

// ---------------------------------------------------------------------------
// WMMA 16-bit fragment load from a [row][K] half buffer (ISA 16x32 layout).
// ---------------------------------------------------------------------------
__device__ __forceinline__ v16h frag_ld(const half_t* base, int row,
                                        int stride_h, int k_off, int lane) {
  union { v16h v; unsigned u[8]; } f;
  const unsigned* p = (const unsigned*)(base + row * stride_h);
  const int khi = (lane & 16) ? 8 : 0;
#pragma unroll
  for (int q = 0; q < 8; ++q) {
    int k = ((q & 3) << 1) + ((q & 4) << 2) + khi + k_off;
    f.u[q] = p[k >> 1];
  }
  return f.v;
}

// ---------------------------------------------------------------------------
// GEMM: C[M,N] = A[M,K] * B[N,K]^T. Tiles fetched by the Tensor Data Mover
// into double-buffered LDS; one wave drives the TDM, all 8 waves compute.
// ---------------------------------------------------------------------------
template <bool HALF_OUT>
__global__ __launch_bounds__(256)
void gemm_wmma(const half_t* __restrict__ A, const half_t* __restrict__ B,
               half_t* __restrict__ Ch, float* __restrict__ Cf,
               int Mtot, int Ntot, int Ktot) {
  constexpr int BM = 64, BN = 128, BK = 32;
  __shared__ alignas(16) half_t As[2][BM * BK];
  __shared__ alignas(16) half_t Bs[2][BN * BK];
  const int tid  = threadIdx.x;
  const int lane = tid & 31;
  const int wave = tid >> 5;
  const int m0 = blockIdx.y * BM;
  const int n0 = blockIdx.x * BN;
  const int m_off  = (wave & 3) * 16;
  const int n_base = (wave >> 2) * 64;
  const int lm = lane & 15;
  const int nk = Ktot / BK;
  v8f acc[4] = {};

  if (wave == 0) {
    tdm_load_2d(lds_off_of(&As[0][0]), A + (size_t)m0 * Ktot, Mtot, Ktot, BM, BK);
    tdm_load_2d(lds_off_of(&Bs[0][0]), B + (size_t)n0 * Ktot, Ntot, Ktot, BN, BK);
  }
  for (int ks = 0; ks < nk; ++ks) {
    const int cur = ks & 1;
    if (wave == 0) {
      if (ks + 1 < nk) {  // prefetch next K tile into the other buffer
        tdm_load_2d(lds_off_of(&As[cur ^ 1][0]),
                    A + (size_t)m0 * Ktot + (ks + 1) * BK, Mtot, Ktot, BM, BK);
        tdm_load_2d(lds_off_of(&Bs[cur ^ 1][0]),
                    B + (size_t)n0 * Ktot + (ks + 1) * BK, Ntot, Ktot, BN, BK);
        __builtin_amdgcn_s_wait_tensorcnt(2);  // previous pair complete
      } else {
        __builtin_amdgcn_s_wait_tensorcnt(0);
      }
    }
    __syncthreads();
    v16h a = frag_ld(As[cur], m_off + lm, BK, 0, lane);
#pragma unroll
    for (int i = 0; i < 4; ++i) {
      v16h b = frag_ld(Bs[cur], n_base + 16 * i + lm, BK, 0, lane);
      acc[i] = __builtin_amdgcn_wmma_f32_16x16x32_f16(
          false, a, false, b, (short)0, acc[i], false, false);
    }
    __syncthreads();
  }

  const int rl = (lane >> 4) << 3;
#pragma unroll
  for (int i = 0; i < 4; ++i) {
    int n = n0 + n_base + 16 * i + lm;
#pragma unroll
    for (int r = 0; r < 8; ++r) {
      int m = m0 + m_off + r + rl;
      if (HALF_OUT) Ch[(size_t)m * Ntot + n] = (half_t)acc[i][r];
      else          Cf[(size_t)m * Ntot + n] = acc[i][r];
    }
  }
}

// ---------------------------------------------------------------------------
// Flash attention, causal, GQA n_rep=4. K tile fetched with async global->LDS
// loads (ASYNCcnt path); V transposed into LDS; P through per-wave LDS strip.
// ---------------------------------------------------------------------------
__global__ __launch_bounds__(128)
void flash_attn(const half_t* __restrict__ Q, const half_t* __restrict__ Kg,
                const half_t* __restrict__ V, half_t* __restrict__ O) {
  __shared__ alignas(16) half_t Ks[64 * HD];    // [keypos][hd]
  __shared__ alignas(16) half_t Vt[HD * 64];    // [hd][keypos]
  __shared__ alignas(16) half_t Pst[4 * 16 * 64];
  const int qb  = blockIdx.x;
  const int h   = blockIdx.y;
  const int kvh = h >> 2;
  const int tid  = threadIdx.x;
  const int lane = tid & 31;
  const int wave = tid >> 5;
  const int m_off = wave * 16;
  const int qrow0 = qb * 64;
  const int lm = lane & 15;
  const int rl = (lane >> 4) << 3;

  union { v16h v; unsigned u[8]; } qa[4];
  {
    const unsigned* qp = (const unsigned*)(
        Q + (size_t)(qrow0 + m_off + lm) * (NH * HD) + h * HD);
    const int khi = (lane & 16) ? 8 : 0;
#pragma unroll
    for (int kk = 0; kk < 4; ++kk)
#pragma unroll
      for (int q = 0; q < 8; ++q) {
        int k = ((q & 3) << 1) + ((q & 4) << 2) + khi + kk * 32;
        qa[kk].u[q] = qp[k >> 1];
      }
  }

  float mrun[8], lrun[8];
  v8f o[8] = {};
#pragma unroll
  for (int r = 0; r < 8; ++r) { mrun[r] = -1e30f; lrun[r] = 0.f; }
  half_t* myP = Pst + wave * 16 * 64;

  for (int kb = 0; kb <= qb; ++kb) {
    const int krow0 = kb * 64;
    // K tile: async global->LDS B128 copies (no VGPR staging)
    for (int i = tid; i < 64 * 16; i += 128) {
      int row = i >> 4, ch = i & 15;
      unsigned ldsa = lds_off_of(Ks + row * HD + ch * 8);
      unsigned voff = (unsigned)((((size_t)(krow0 + row)) * (NKV * HD) +
                                  kvh * HD + ch * 8) * 2);
      asm volatile("global_load_async_to_lds_b128 %0, %1, %2"
                   :: "v"(ldsa), "v"(voff), "s"(Kg) : "memory");
    }
    // V tile, transposed (regular path)
    for (int i = tid; i < 64 * 64; i += 128) {
      int row = i & 63;
      int c2 = (i >> 6) << 1;
      union { unsigned u; half_t h2[2]; } cv;
      cv.u = *(const unsigned*)(V + (size_t)(krow0 + row) * (NKV * HD) +
                                kvh * HD + c2);
      Vt[c2 * 64 + row]       = cv.h2[0];
      Vt[(c2 + 1) * 64 + row] = cv.h2[1];
    }
    asm volatile("s_wait_asynccnt 0" ::: "memory");
    __syncthreads();

    // S = Q * K^T
    v8f s[4];
#pragma unroll
    for (int j = 0; j < 4; ++j) {
      v8f z = {};
#pragma unroll
      for (int kk = 0; kk < 4; ++kk) {
        v16h b = frag_ld(Ks, 16 * j + lm, HD, kk * 32, lane);
        z = __builtin_amdgcn_wmma_f32_16x16x32_f16(
            false, qa[kk].v, false, b, (short)0, z, false, false);
      }
      s[j] = z;
    }
    const float scale = 0.08838834764831845f;
    const bool diag = (kb == qb);
#pragma unroll
    for (int j = 0; j < 4; ++j) {
      int col = 16 * j + lm;
#pragma unroll
      for (int r = 0; r < 8; ++r) {
        float v2 = s[j][r] * scale;
        if (diag && col > (m_off + r + rl)) v2 = -1e30f;
        s[j][r] = v2;
      }
    }

    // online softmax
#pragma unroll
    for (int r = 0; r < 8; ++r) {
      float mx = s[0][r];
#pragma unroll
      for (int j = 1; j < 4; ++j) mx = fmaxf(mx, s[j][r]);
#pragma unroll
      for (int d = 1; d < 16; d <<= 1) mx = fmaxf(mx, __shfl_xor(mx, d, 32));
      float mnew = fmaxf(mrun[r], mx);
      float corr = __expf(mrun[r] - mnew);
      float lsum = 0.f;
#pragma unroll
      for (int j = 0; j < 4; ++j) {
        float p = __expf(s[j][r] - mnew);
        s[j][r] = p;
        lsum += p;
      }
#pragma unroll
      for (int d = 1; d < 16; d <<= 1) lsum += __shfl_xor(lsum, d, 32);
      lrun[r] = lrun[r] * corr + lsum;
      mrun[r] = mnew;
#pragma unroll
      for (int n = 0; n < 8; ++n) o[n][r] *= corr;
    }

    // P: C-layout -> A-layout via per-wave LDS strip
#pragma unroll
    for (int j = 0; j < 4; ++j) {
      int col = 16 * j + lm;
#pragma unroll
      for (int r = 0; r < 8; ++r)
        myP[(r + rl) * 64 + col] = (half_t)s[j][r];
    }
    __syncthreads();

    // O += P * V
#pragma unroll
    for (int ks = 0; ks < 2; ++ks) {
      v16h a = frag_ld(myP, lm, 64, ks * 32, lane);
#pragma unroll
      for (int n = 0; n < 8; ++n) {
        v16h b = frag_ld(Vt, 16 * n + lm, 64, ks * 32, lane);
        o[n] = __builtin_amdgcn_wmma_f32_16x16x32_f16(
            false, a, false, b, (short)0, o[n], false, false);
      }
    }
    __syncthreads();
  }

#pragma unroll
  for (int r = 0; r < 8; ++r) {
    float inv = 1.0f / lrun[r];
#pragma unroll
    for (int n = 0; n < 8; ++n) {
      O[(size_t)(qrow0 + m_off + r + rl) * (NH * HD) + h * HD + 16 * n + lm] =
          (half_t)(o[n][r] * inv);
    }
  }
}

// ---------------------------------------------------------------------------
extern "C" void kernel_launch(void* const* d_in, const int* in_sizes, int n_in,
                              void* d_out, int out_size, void* d_ws,
                              size_t ws_size, hipStream_t stream) {
  (void)in_sizes; (void)n_in; (void)out_size; (void)ws_size;
  const float* x  = (const float*)d_in[0];
  const float* wq = (const float*)d_in[1];
  const float* wk = (const float*)d_in[2];
  const float* wv = (const float*)d_in[3];
  const float* wo = (const float*)d_in[4];
  const float* fc = (const float*)d_in[5];
  const float* fs = (const float*)d_in[6];
  float* out = (float*)d_out;

  char* p = (char*)d_ws;
  half_t* wq_h = (half_t*)p; p += (size_t)DIM * DIM * 2;
  half_t* wk_h = (half_t*)p; p += (size_t)NKV * HD * DIM * 2;
  half_t* wv_h = (half_t*)p; p += (size_t)NKV * HD * DIM * 2;
  half_t* wo_h = (half_t*)p; p += (size_t)DIM * DIM * 2;
  half_t* x_h  = (half_t*)p; p += (size_t)SEQ * DIM * 2;
  half_t* q_h  = (half_t*)p; p += (size_t)SEQ * NH * HD * 2;
  half_t* k_h  = (half_t*)p; p += (size_t)SEQ * NKV * HD * 2;
  half_t* v_h  = (half_t*)p; p += (size_t)SEQ * NKV * HD * 2;
  half_t* a_h  = (half_t*)p; p += (size_t)SEQ * NH * HD * 2;

  bfp_quant_to_f16<<<64, 256, 0, stream>>>(wq, wq_h, DIM, DIM);
  bfp_quant_to_f16<<<64, 256, 0, stream>>>(wk, wk_h, NKV * HD, DIM);
  bfp_quant_to_f16<<<64, 256, 0, stream>>>(wv, wv_h, NKV * HD, DIM);
  bfp_quant_to_f16<<<64, 256, 0, stream>>>(wo, wo_h, DIM, DIM);
  f32_to_f16<<<(SEQ * DIM) / 1024, 256, 0, stream>>>(x, x_h, SEQ * DIM);

  gemm_wmma<true><<<dim3(DIM / 128, SEQ / 64), 256, 0, stream>>>(
      x_h, wq_h, q_h, nullptr, SEQ, DIM, DIM);
  gemm_wmma<true><<<dim3((NKV * HD) / 128, SEQ / 64), 256, 0, stream>>>(
      x_h, wk_h, k_h, nullptr, SEQ, NKV * HD, DIM);
  gemm_wmma<true><<<dim3((NKV * HD) / 128, SEQ / 64), 256, 0, stream>>>(
      x_h, wv_h, v_h, nullptr, SEQ, NKV * HD, DIM);

  rope_kernel<<<(SEQ * NH * (HD / 2)) / 256, 256, 0, stream>>>(q_h, fc, fs, NH);
  rope_kernel<<<(SEQ * NKV * (HD / 2)) / 256, 256, 0, stream>>>(k_h, fc, fs, NKV);

  flash_attn<<<dim3(SEQ / 64, NH), 128, 0, stream>>>(q_h, k_h, v_h, a_h);

  gemm_wmma<false><<<dim3(DIM / 128, SEQ / 64), 256, 0, stream>>>(
      a_h, wo_h, nullptr, out, SEQ, DIM, DIM);
}
